// FCN_mix_45114336477857
// MI455X (gfx1250) — 8-bit matrix operand order repaired and checked by reading, compiled, not run on hardware
//
#include <hip/hip_runtime.h>

typedef __attribute__((ext_vector_type(8))) int v8i;

#define WAVES_PER_BLOCK 8
#define THREADS 256
#define WQ_INTS 2816

__device__ __forceinline__ int q8s(float x, bool high) {
  if (high) {
    float r = rintf(x * 16.0f);
    r = fminf(fmaxf(r, -128.0f), 127.0f);
    return (int)r;
  } else {
    float r = rintf(x * 8.0f);
    r = fminf(fmaxf(r, -8.0f), 7.0f);
    return 2 * (int)r;
  }
}

__device__ __forceinline__ int pack4(int b0, int b1, int b2, int b3) {
  return (b0 & 255) | ((b1 & 255) << 8) | ((b2 & 255) << 16) | ((b3 & 255) << 24);
}

__device__ __forceinline__ int qbias16(float b) {
  float r = rintf(b * 16.0f);
  r = fminf(fmaxf(r, -128.0f), 127.0f);
  return ((int)r) << 4;
}

__global__ __launch_bounds__(THREADS) void fcn_mix_kernel(
    const float* __restrict__ x,
    const float* __restrict__ W1, const float* __restrict__ b1,
    const float* __restrict__ W2, const float* __restrict__ W3,
    const float* __restrict__ W4, const float* __restrict__ W5,
    const float* __restrict__ b5,
    float* __restrict__ out, int nTiles) {
  __shared__ int wq[WQ_INTS];
  __shared__ int act[WAVES_PER_BLOCK][2][16 * 16];

  const int tid = threadIdx.x;
  const int lane = tid & 31;
  const int waveInBlock = tid >> 5;
  const int nl = lane & 15;
  const int h = lane >> 4;

  auto stage = [&](const float* __restrict__ W, int nOut, int kIn, int baseInt, int rowsPad) {
    for (int idx = tid; idx < rowsPad * 16; idx += THREADS) {
      int n = idx >> 4;
      int k0 = (idx & 15) * 4;
      int packed = 0;
      if (n < nOut) {
        int q[4];
#pragma unroll
        for (int j = 0; j < 4; ++j) {
          int k = k0 + j;
          q[j] = (k < kIn) ? q8s(W[n * kIn + k], (k & 15) < 8) : 0;
        }
        packed = pack4(q[0], q[1], q[2], q[3]);
      }
      wq[baseInt + idx] = packed;
    }
  };
  stage(W1, 64, 16, 0, 64);
  stage(W2, 32, 64, 1024, 32);
  stage(W3, 32, 32, 1536, 32);
  stage(W4, 32, 32, 2048, 32);
  stage(W5, 5, 32, 2560, 16);
  __syncthreads();

  auto loadB = [&](int baseInt, int nOff) -> v8i {
    const int2* p = (const int2*)(wq + baseInt + (nOff + nl) * 16 + h * 2);
    int2 q0 = p[0], q1 = p[2], q2 = p[4], q3 = p[6];
    v8i b;
    b[0] = q0.x; b[1] = q0.y; b[2] = q1.x; b[3] = q1.y;
    b[4] = q2.x; b[5] = q2.y; b[6] = q3.x; b[7] = q3.y;
    return b;
  };
  v8i B1f[4], B2f[2], B3f[2], B4f[2], B5f;
  B1f[0] = loadB(0, 0);  B1f[1] = loadB(0, 16);  B1f[2] = loadB(0, 32);  B1f[3] = loadB(0, 48);
  B2f[0] = loadB(1024, 0); B2f[1] = loadB(1024, 16);
  B3f[0] = loadB(1536, 0); B3f[1] = loadB(1536, 16);
  B4f[0] = loadB(2048, 0); B4f[1] = loadB(2048, 16);
  B5f = loadB(2560, 0);

  int b1q[4];
#pragma unroll
  for (int t = 0; t < 4; ++t) b1q[t] = qbias16(b1[t * 16 + nl]);
  float b5v = (nl < 5) ? (float)qbias16(b5[nl]) * (1.0f / 256.0f) : 0.0f;

  int* aldsA = &act[waveInBlock][0][0];
  int* aldsB = &act[waveInBlock][1][0];

  const int highMask = (nl < 8) ? -1 : 0;
  const bool highA = (h == 0);
  const float qsc = highA ? 16.0f : 8.0f;
  const float qlo = highA ? -128.0f : -8.0f;
  const float qhi = highA ? 127.0f : 7.0f;

  auto loadA = [&](const int* alds) -> v8i {
    const int2* p = (const int2*)(alds + nl * 16 + h * 2);
    int2 p0 = p[0], p1 = p[2], p2 = p[4], p3 = p[6];
    v8i a;
    a[0] = p0.x; a[1] = p0.y; a[2] = p1.x; a[3] = p1.y;
    a[4] = p2.x; a[5] = p2.y; a[6] = p3.x; a[7] = p3.y;
    return a;
  };

  auto quantX = [&](int tl) -> v8i {
    const float* xrow = x + (size_t)tl * 256;
    const float4* xv = reinterpret_cast<const float4*>(xrow + nl * 16 + h * 8);
    float4 x0 = xv[0];
    float4 x1 = xv[1];
    auto q8 = [&](float xx) -> int {
      float r = rintf(xx * qsc);
      r = fminf(fmaxf(r, qlo), qhi);
      return ((int)r) << h;
    };
    v8i a;
    a[0] = pack4(q8(x0.x), q8(x0.y), q8(x0.z), q8(x0.w));
    a[1] = pack4(q8(x1.x), q8(x1.y), q8(x1.z), q8(x1.w));
    a[2] = 0; a[3] = 0; a[4] = 0; a[5] = 0; a[6] = 0; a[7] = 0;
    return a;
  };

  auto postStore = [&](int* alds, v8i acc, int bias16, int nOff) {
    char* ab = (char*)alds;
    const int n = nOff + nl;
#pragma unroll
    for (int r = 0; r < 8; ++r) {
      int total = acc[r] + bias16;
      int t = total / 256;
      int w8 = (int)(signed char)(t & 255);
      int v = w8 > 0 ? w8 : 0;
      int hi = v << 4;
      hi = hi > 127 ? 127 : hi;
      int lo = (v != 0) ? 14 : 0;
      int aq = lo + ((hi - lo) & highMask);
      ab[(r + 8 * h) * 64 + n] = (signed char)aq;
    }
  };

  auto storeOut = [&](int tl, v8i acc) {
#pragma unroll
    for (int r = 0; r < 8; ++r) {
      float y = (float)acc[r] * (1.0f / 256.0f) + b5v;
      y = fmaxf(y, 0.0f);
      out[(size_t)(tl * 16 + r + 8 * h) * 5 + nl] = y;
    }
  };

  const v8i zero = {0, 0, 0, 0, 0, 0, 0, 0};
  const int waveGlobal = blockIdx.x * WAVES_PER_BLOCK + waveInBlock;
  const int waveCount = gridDim.x * WAVES_PER_BLOCK;

  for (int tile = waveGlobal; tile < nTiles; tile += 2 * waveCount) {
    int tileB = tile + waveCount;
    if (tileB >= nTiles) tileB = tile;

    {
      v8i aA = quantX(tile);
      v8i aB = quantX(tileB);
#pragma unroll
      for (int t = 0; t < 4; ++t) {
        v8i accA = __builtin_amdgcn_wmma_i32_16x16x64_iu8(true, aA, true, B1f[t], zero, false, false);
        v8i accB = __builtin_amdgcn_wmma_i32_16x16x64_iu8(true, aB, true, B1f[t], zero, false, false);
        postStore(aldsA, accA, b1q[t], t * 16);
        postStore(aldsB, accB, b1q[t], t * 16);
      }
    }
    asm volatile("s_wait_dscnt 0" ::: "memory");

    {
      v8i aA = loadA(aldsA);
      v8i aB = loadA(aldsB);
#pragma unroll
      for (int t = 0; t < 2; ++t) {
        v8i accA = __builtin_amdgcn_wmma_i32_16x16x64_iu8(true, aA, true, B2f[t], zero, false, false);
        v8i accB = __builtin_amdgcn_wmma_i32_16x16x64_iu8(true, aB, true, B2f[t], zero, false, false);
        postStore(aldsA, accA, 0, t * 16);
        postStore(aldsB, accB, 0, t * 16);
      }
    }
    asm volatile("s_wait_dscnt 0" ::: "memory");

    {
      v8i aA = loadA(aldsA);
      v8i aB = loadA(aldsB);
#pragma unroll
      for (int t = 0; t < 2; ++t) {
        v8i accA = __builtin_amdgcn_wmma_i32_16x16x64_iu8(true, aA, true, B3f[t], zero, false, false);
        v8i accB = __builtin_amdgcn_wmma_i32_16x16x64_iu8(true, aB, true, B3f[t], zero, false, false);
        postStore(aldsA, accA, 0, t * 16);
        postStore(aldsB, accB, 0, t * 16);
      }
    }
    asm volatile("s_wait_dscnt 0" ::: "memory");

    {
      v8i aA = loadA(aldsA);
      v8i aB = loadA(aldsB);
#pragma unroll
      for (int t = 0; t < 2; ++t) {
        v8i accA = __builtin_amdgcn_wmma_i32_16x16x64_iu8(true, aA, true, B4f[t], zero, false, false);
        v8i accB = __builtin_amdgcn_wmma_i32_16x16x64_iu8(true, aB, true, B4f[t], zero, false, false);
        postStore(aldsA, accA, 0, t * 16);
        postStore(aldsB, accB, 0, t * 16);
      }
    }
    asm volatile("s_wait_dscnt 0" ::: "memory");

    {
      v8i aA = loadA(aldsA);
      v8i aB = loadA(aldsB);
      v8i accA = __builtin_amdgcn_wmma_i32_16x16x64_iu8(true, aA, true, B5f, zero, false, false);
      v8i accB = __builtin_amdgcn_wmma_i32_16x16x64_iu8(true, aB, true, B5f, zero, false, false);
      if (nl < 5) {
        storeOut(tile, accA);
        storeOut(tileB, accB);
      }
    }
  }
}

extern "C" void kernel_launch(void* const* d_in, const int* in_sizes, int n_in,
                              void* d_out, int out_size, void* d_ws, size_t ws_size,
                              hipStream_t stream) {
  const float* x  = (const float*)d_in[0];
  const float* W1 = (const float*)d_in[1];
  const float* b1 = (const float*)d_in[2];
  const float* W2 = (const float*)d_in[3];
  const float* W3 = (const float*)d_in[4];
  const float* W4 = (const float*)d_in[5];
  const float* W5 = (const float*)d_in[6];
  const float* b5 = (const float*)d_in[7];
  float* out = (float*)d_out;

  int B = in_sizes[0] / 16;
  int nTiles = B / 16;
  int blocks = (nTiles + WAVES_PER_BLOCK - 1) / WAVES_PER_BLOCK;
  if (blocks > 1024) blocks = 1024;
  if (blocks < 1) blocks = 1;

  fcn_mix_kernel<<<blocks, THREADS, 0, stream>>>(x, W1, b1, W2, W3, W4, W5, b5, out, nTiles);
}
